// VarWhisperAttention_73280732004946
// MI455X (gfx1250) — matrix operand order repaired and checked by reading, not compiled, not run on hardware
//
#include <hip/hip_runtime.h>

typedef __attribute__((ext_vector_type(16))) _Float16 v16h;
typedef __attribute__((ext_vector_type(8)))  float    v8f;

#define BSZ   2
#define SEQL  2048
#define NH    16
#define HD    64
#define WIN   128
#define MASKM (-1e9f)

// ---- DPP16 ROW_XMASK xor-butterfly within each 16-lane half (pure VALU, no LDS)
template <int CTRL>
__device__ __forceinline__ float dpp_xor(float x) {
    const int xi = __builtin_bit_cast(int, x);
    const int yi = __builtin_amdgcn_update_dpp(xi, xi, CTRL, 0xF, 0xF, true);
    return __builtin_bit_cast(float, yi);
}
__device__ __forceinline__ float half_max16(float x) {   // max over 16-lane half
    x = fmaxf(x, dpp_xor<0x161>(x));   // ROW_XMASK:1
    x = fmaxf(x, dpp_xor<0x162>(x));   // ROW_XMASK:2
    x = fmaxf(x, dpp_xor<0x164>(x));   // ROW_XMASK:4
    x = fmaxf(x, dpp_xor<0x168>(x));   // ROW_XMASK:8
    return x;
}
__device__ __forceinline__ float half_sum16(float x) {   // sum over 16-lane half
    x += dpp_xor<0x161>(x);
    x += dpp_xor<0x162>(x);
    x += dpp_xor<0x164>(x);
    x += dpp_xor<0x168>(x);
    return x;
}

// ---------------- Pass 1: RoPE + 1/sqrt(D) scale + pack to f16
// Q rowmajor [BH][S][64] (pre-scaled), K transposed [BH][64][S], V rowmajor [BH][S][64]
__global__ __launch_bounds__(64)
void rope_pack_kernel(const float* __restrict__ qkv,
                      const int*   __restrict__ pos_ids,
                      _Float16* __restrict__ Qh,
                      _Float16* __restrict__ Kt,
                      _Float16* __restrict__ Vh) {
    const int row = blockIdx.x;            // b*S*H + s*H + h
    const int d   = threadIdx.x;           // 0..63
    const int h   = row % NH;
    const int s   = (row / NH) % SEQL;
    const int b   = row / (NH * SEQL);

    const int   pos = pos_ids[b * SEQL + s];
    const int   t   = d & 31;
    const float inv_freq = __expf(-(float)t * (9.210340371976184f / 32.0f)); // 10000^(-t/32)
    const float fr = (float)pos * inv_freq;
    const float c  = cosf(fr);
    const float sn = sinf(fr);

    const size_t base = ((size_t)(b * SEQL + s) * 3) * (NH * HD) + (size_t)h * HD;
    const int partner = (d < 32) ? d + 32 : d - 32;
    const float sgn   = (d < 32) ? -1.0f : 1.0f;

    const float q  = qkv[base + d];
    const float q2 = qkv[base + partner];
    const float k  = qkv[base + (size_t)NH * HD + d];
    const float k2 = qkv[base + (size_t)NH * HD + partner];
    const float v  = qkv[base + (size_t)2 * NH * HD + d];

    const float qr = (q * c + sgn * q2 * sn) * 0.125f;   // fold softmax scale into Q
    const float kr = k * c + sgn * k2 * sn;

    const int bh = b * NH + h;
    Qh[((size_t)bh * SEQL + s) * HD + d]   = (_Float16)qr;
    Kt[((size_t)bh * HD   + d) * SEQL + s] = (_Float16)kr;
    Vh[((size_t)bh * SEQL + s) * HD + d]   = (_Float16)v;
}

// ---------------- Pass 2: flash attention, one wave32 per 16-row query tile
union Frag16 { v16h v; uint4 u[2]; };

__global__ __launch_bounds__(32)
void attn_kernel(const _Float16* __restrict__ Qh,
                 const _Float16* __restrict__ Kt,
                 const _Float16* __restrict__ Vh,
                 float* __restrict__ out) {
    __shared__ __align__(32) _Float16 Pbuf[16 * 32];   // P tile, row-major 16x32

    const int lane = threadIdx.x;
    const int hlf  = lane >> 4;        // 0/1
    const int m16  = lane & 15;
    const int i0   = blockIdx.x * 16;  // query tile start
    const int bh   = blockIdx.y;
    const int b    = bh / NH;
    const int h    = bh % NH;

    // Q A-fragments (persist across key loop): chunk c covers dims c*32..c*32+31
    Frag16 qa[2];
    {
        const _Float16* qrow = Qh + ((size_t)bh * SEQL + (i0 + m16)) * HD;
#pragma unroll
        for (int c = 0; c < 2; ++c) {
            qa[c].u[0] = *(const uint4*)(qrow + c * 32 + hlf * 8);
            qa[c].u[1] = *(const uint4*)(qrow + c * 32 + 16 + hlf * 8);
        }
    }

    v8f acc[4];
#pragma unroll
    for (int n = 0; n < 4; ++n) acc[n] = {};
    float mrow[8], lrow[8];                 // lrow = per-LANE partial sums
#pragma unroll
    for (int r = 0; r < 8; ++r) { mrow[r] = -INFINITY; lrow[r] = 0.0f; }

    // per-row valid key range [lo_r, hi_r] (hoisted out of the chunk loop)
    int lo[8], hi[8];
#pragma unroll
    for (int r = 0; r < 8; ++r) {
        const int i = i0 + (hlf ? r + 8 : r);
        lo[r] = (i - WIN > 0) ? i - WIN : 0;
        hi[r] = (i + WIN < SEQL - 1) ? i + WIN : SEQL - 1;
    }

    const _Float16* Kbase = Kt + (size_t)bh * HD * SEQL;
    const _Float16* Vbase = Vh + (size_t)bh * SEQL * HD;

    for (int ch = 0; ch < 9; ++ch) {
        const int j0 = i0 - WIN + ch * 32;   // 32 keys: two 16-key tiles

        // prefetch next chunk's K columns and V row (speculative, L2-resident anyway)
        {
            const int jsn = ((j0 + 32) >= 0 && (j0 + 48) <= SEQL) ? (j0 + 32) : 0;
            __builtin_prefetch(Kbase + (size_t)lane * SEQL + jsn, 0, 0);
            __builtin_prefetch(Kbase + (size_t)(32 + lane) * SEQL + jsn, 0, 0);
            int jn = j0 + 32 + lane;
            jn = (jn >= 0 && jn < SEQL) ? jn : 0;
            __builtin_prefetch(Vbase + (size_t)jn * HD, 0, 0);
        }

        // ---- S = Q * K^T  (two 16x16 C tiles)
        v8f c0 = {}, c1 = {};
#pragma unroll
        for (int t = 0; t < 2; ++t) {
            const int js  = j0 + 16 * t;
            const int jsc = (js >= 0 && js + 16 <= SEQL) ? js : 0;  // tile fully in/out
            Frag16 kb0, kb1;
#pragma unroll
            for (int s = 0; s < 16; ++s) {
                const int kd = (s < 8) ? hlf * 8 + s : 8 + hlf * 8 + s;
                kb0.v[s] = Kbase[(size_t)kd * SEQL + jsc + m16];
                kb1.v[s] = Kbase[(size_t)(32 + kd) * SEQL + jsc + m16];
            }
            if (t == 0) {
                c0 = __builtin_amdgcn_wmma_f32_16x16x32_f16(false, qa[0].v, false, kb0.v, (short)0, c0, false, false);
                c0 = __builtin_amdgcn_wmma_f32_16x16x32_f16(false, qa[1].v, false, kb1.v, (short)0, c0, false, false);
            } else {
                c1 = __builtin_amdgcn_wmma_f32_16x16x32_f16(false, qa[0].v, false, kb0.v, (short)0, c1, false, false);
                c1 = __builtin_amdgcn_wmma_f32_16x16x32_f16(false, qa[1].v, false, kb1.v, (short)0, c1, false, false);
            }
        }

        // ---- online softmax (C layout: VGPR r = row M / M+8 per half, N = lane&15)
        const int jA = j0 + m16;
        const int jB = j0 + 16 + m16;
#pragma unroll
        for (int r = 0; r < 8; ++r) {
            const bool vA = (jA >= lo[r]) && (jA <= hi[r]);
            const bool vB = (jB >= lo[r]) && (jB <= hi[r]);
            const float s0 = vA ? c0[r] : MASKM;   // select is exact after exp underflow
            const float s1 = vB ? c1[r] : MASKM;

            const float mx   = half_max16(fmaxf(s0, s1));
            const float mnew = fmaxf(mrow[r], mx);
            const float corr = __expf(mrow[r] - mnew);
            mrow[r] = mnew;

            const float p0 = __expf(s0 - mnew);
            const float p1 = __expf(s1 - mnew);
            lrow[r] = lrow[r] * corr + (p0 + p1);   // per-lane partial row sum
#pragma unroll
            for (int n = 0; n < 4; ++n) acc[n][r] *= corr;

            const int prow = hlf ? r + 8 : r;
            Pbuf[prow * 32 + m16]      = (_Float16)p0;
            Pbuf[prow * 32 + 16 + m16] = (_Float16)p1;
        }
        __syncthreads();   // single-wave WG: cheap, orders LDS

        // ---- reload P in A-layout
        Frag16 pa;
        pa.u[0] = *(const uint4*)(&Pbuf[m16 * 32 + hlf * 8]);
        pa.u[1] = *(const uint4*)(&Pbuf[m16 * 32 + 16 + hlf * 8]);

        // ---- O += P * V   (B: lane%16 = dim, elements = keys in table order)
#pragma unroll
        for (int n = 0; n < 4; ++n) {
            Frag16 vb;
#pragma unroll
            for (int s = 0; s < 16; ++s) {
                const int kk  = (s < 8) ? hlf * 8 + s : 8 + hlf * 8 + s;
                const int jv  = j0 + kk;
                const int jvc = (jv >= 0 && jv < SEQL) ? jv : 0;
                vb.v[s] = Vbase[(size_t)jvc * HD + n * 16 + m16];
            }
            acc[n] = __builtin_amdgcn_wmma_f32_16x16x32_f16(false, pa.v, false, vb.v, (short)0, acc[n], false, false);
        }
        __syncthreads();   // protect Pbuf WAR before next chunk
    }

    // ---- single deferred row-sum reduction, then normalize + store
#pragma unroll
    for (int r = 0; r < 8; ++r) {
        const int   i   = i0 + (hlf ? r + 8 : r);
        const float inv = 1.0f / half_sum16(lrow[r]);
#pragma unroll
        for (int n = 0; n < 4; ++n) {
            const int d = n * 16 + m16;
            out[((size_t)(b * SEQL + i)) * (NH * HD) + h * HD + d] = acc[n][r] * inv;
        }
    }
}

extern "C" void kernel_launch(void* const* d_in, const int* in_sizes, int n_in,
                              void* d_out, int out_size, void* d_ws, size_t ws_size,
                              hipStream_t stream) {
    const float* qkv = (const float*)d_in[0];
    // d_in[1] attention_mask (all zeros, unused by reference math)
    // d_in[2] sliding_window_mask (computed analytically)
    const int* pos = (const int*)d_in[3];

    const size_t per = (size_t)BSZ * NH * SEQL * HD;   // elements per f16 tensor
    _Float16* Qh = (_Float16*)d_ws;
    _Float16* Kt = Qh + per;
    _Float16* Vh = Kt + per;
    float* out = (float*)d_out;

    rope_pack_kernel<<<BSZ * SEQL * NH, 64, 0, stream>>>(qkv, pos, Qh, Kt, Vh);

    dim3 grid(SEQL / 16, BSZ * NH);
    attn_kernel<<<grid, 32, 0, stream>>>(Qh, Kt, Vh, out);
}
